// GNN_18786186953251
// MI455X (gfx1250) — compile-verified, matrix operand order checked by reading, not run on hardware
//
#include <hip/hip_runtime.h>
#include <hip/hip_bf16.h>

// ---------------------------------------------------------------------------
// Types for WMMA fragments (gfx1250, wave32)
// ---------------------------------------------------------------------------
typedef __attribute__((ext_vector_type(16))) _Float16 v16h;
typedef __attribute__((ext_vector_type(8)))  _Float16 v8h;   // 16 bytes
typedef __attribute__((ext_vector_type(8)))  float    v8f;

union Frag16 { v16h v; v8h h[2]; };

#define BB 32
#define NN 512
#define JJ 6
#define NF 20
#define NL 20
#define ROWS (BB * NN)      // 16384
#define HTF  32             // padded feature rows in transposed H
#define NBLK 64             // blocks in fc_bn / bn_normalize

// ---------------------------------------------------------------------------
// Kernel 1: repack W f32 [B][N][M][J] -> f16 [B][J][N][M]  (one HBM pass)
// One thread per (b,n,m): reads 6 contiguous f32, writes 6 planes coalesced.
// ---------------------------------------------------------------------------
__global__ void __launch_bounds__(256)
repack_w(const float* __restrict__ W, _Float16* __restrict__ Wh) {
    size_t i = (size_t)blockIdx.x * blockDim.x + threadIdx.x;   // b*N*M range
    int m = (int)(i & 511);
    size_t t = i >> 9;
    int n = (int)(t & 511);
    int b = (int)(t >> 9);
    const float* src = W + i * 6;
#pragma unroll
    for (int j = 0; j < 6; ++j) {
        Wh[(((size_t)b * JJ + j) * NN + n) * NN + m] = (_Float16)src[j];
    }
}

// ---------------------------------------------------------------------------
// Kernel 2: zero the padding rows f=20..31 of the transposed activations
// ---------------------------------------------------------------------------
__global__ void __launch_bounds__(256)
zero_pad(_Float16* __restrict__ Ht) {
    int i = blockIdx.x * blockDim.x + threadIdx.x;   // over B*12*512
    int n = i & 511;
    int t = i >> 9;
    int f = NF + (t % (HTF - NF));
    int b = t / (HTF - NF);
    Ht[((size_t)b * HTF + f) * NN + n] = (_Float16)0.f;
}

// ---------------------------------------------------------------------------
// Kernel 3: layer-0 gmul (F=1): y[b,n,j] = sum_m Wh[b][j][n][m] * x[b][m]
// ---------------------------------------------------------------------------
__global__ void __launch_bounds__(256)
gmul0(const _Float16* __restrict__ Wh, const float* __restrict__ x,
      float* __restrict__ Y) {
    int i = blockIdx.x * blockDim.x + threadIdx.x;   // (b*6 + j)*512 + n
    int n = i & 511;
    int t = i >> 9;
    int j = t % 6;
    int b = t / 6;
    const _Float16* row = Wh + (((size_t)b * JJ + j) * NN + n) * NN;
    const float* xb = x + (size_t)b * NN;
    float s = 0.f;
    for (int m = 0; m < NN; ++m) s += (float)row[m] * xb[m];
    Y[((size_t)b * NN + n) * JJ + j] = s;
}

// ---------------------------------------------------------------------------
// Kernel 4: WMMA GEMM.  For each (b, j):  Hm[b][:, j*20 + f] = Wh_bj @ H_b
//   Wh:  f16 [B][J][512][512]  (A matrix, row-major M x K)
//   Ht:  f16 [B][32][512]      (B matrix stored [N][K]; rows 20..31 are zero)
//   Hm:  f32 [B][512][120]
// Block = 128 threads = 4 waves; each wave owns a 16(M) x 32(F) tile.
// Grid  = B * J * (512/64) = 1536.
// Fragment addressing per ISA 7.12.2 (16-bit, wave32):
//   lanes 0-15  : row (lane&15), K offsets {0..7} in v0..3, {16..23} in v4..7
//   lanes 16-31 : row (lane&15), K offsets {8..15} in v0..3, {24..31} in v4..7
// ---------------------------------------------------------------------------
__global__ void __launch_bounds__(128)
gemm_wmma(const _Float16* __restrict__ Wh, const _Float16* __restrict__ Ht,
          float* __restrict__ Hm) {
    const int lane  = threadIdx.x & 31;
    const int wave  = threadIdx.x >> 5;
    int bid = blockIdx.x;
    const int mg = bid & 7;  bid >>= 3;          // m-tile group (64 rows)
    const int j  = bid % 6;
    const int b  = bid / 6;

    const int m0    = mg * 64 + wave * 16;
    const int lrow  = lane & 15;
    const int lhalf = lane >> 4;                  // 0 or 1

    const _Float16* Ab  = Wh + (((size_t)b * JJ + j) * NN + (m0 + lrow)) * NN + lhalf * 8;
    const _Float16* B0b = Ht + ((size_t)b * HTF + lrow)      * NN + lhalf * 8;
    const _Float16* B1b = Ht + ((size_t)b * HTF + 16 + lrow) * NN + lhalf * 8;

    v8f c0 = {};
    v8f c1 = {};
#pragma unroll 4
    for (int k0 = 0; k0 < NN; k0 += 32) {
        Frag16 a, p, q;
        a.h[0] = *(const v8h*)(Ab  + k0);
        a.h[1] = *(const v8h*)(Ab  + k0 + 16);
        p.h[0] = *(const v8h*)(B0b + k0);
        p.h[1] = *(const v8h*)(B0b + k0 + 16);
        q.h[0] = *(const v8h*)(B1b + k0);
        q.h[1] = *(const v8h*)(B1b + k0 + 16);
        c0 = __builtin_amdgcn_wmma_f32_16x16x32_f16(false, a.v, false, p.v,
                                                    (short)0, c0, false, false);
        c1 = __builtin_amdgcn_wmma_f32_16x16x32_f16(false, a.v, false, q.v,
                                                    (short)0, c1, false, false);
    }

    // D layout: VGPR r -> row (lhalf*8 + r), column = lrow (+16 for c1 tile).
    float* out = Hm + (size_t)b * NN * (JJ * NF);
    const int mrow = m0 + lhalf * 8;
#pragma unroll
    for (int r = 0; r < 8; ++r) {
        out[(size_t)(mrow + r) * (JJ * NF) + j * NF + lrow] = c0[r];
        if (lrow < NF - 16)  // only f=16..19 of the second tile are real
            out[(size_t)(mrow + r) * (JJ * NF) + j * NF + 16 + lrow] = c1[r];
    }
}

// ---------------------------------------------------------------------------
// Kernel 5: FC (d -> 10 relu | 10 linear) + per-block BN partial sums.
// One thread per row. Deterministic block reduction (shuffle + LDS).
//   Hin:     f32 [16384][d]
//   Hc:      f32 [16384][20]
//   partial: f32 [NBLK][40]  (sum[20], sumsq[20])
// ---------------------------------------------------------------------------
__global__ void __launch_bounds__(256)
fc_bn(const float* __restrict__ Hin, int d,
      const float* __restrict__ w1, const float* __restrict__ b1,
      const float* __restrict__ w2, const float* __restrict__ b2,
      float* __restrict__ Hc, float* __restrict__ partial) {
    const int row  = blockIdx.x * blockDim.x + threadIdx.x;
    const int lane = threadIdx.x & 31;
    const int wid  = threadIdx.x >> 5;

    float acc[20];
#pragma unroll
    for (int c = 0; c < 10; ++c) acc[c]      = b1[c];
#pragma unroll
    for (int c = 0; c < 10; ++c) acc[10 + c] = b2[c];

    const float* hr = Hin + (size_t)row * d;
    for (int k = 0; k < d; ++k) {
        float hv = hr[k];
#pragma unroll
        for (int c = 0; c < 10; ++c) acc[c]      += hv * w1[c * d + k];   // uniform -> scalar loads
#pragma unroll
        for (int c = 0; c < 10; ++c) acc[10 + c] += hv * w2[c * d + k];
    }
#pragma unroll
    for (int c = 0; c < 10; ++c) acc[c] = fmaxf(acc[c], 0.f);

    float* hcr = Hc + (size_t)row * NF;
#pragma unroll
    for (int c = 0; c < NF; ++c) hcr[c] = acc[c];

    // deterministic reduction: wave32 shuffles, then 8 warps via LDS
    __shared__ float red[8][2 * NF];
#pragma unroll
    for (int c = 0; c < NF; ++c) {
        float s = acc[c];
        float q = acc[c] * acc[c];
#pragma unroll
        for (int o = 16; o > 0; o >>= 1) {
            s += __shfl_down(s, o, 32);
            q += __shfl_down(q, o, 32);
        }
        if (lane == 0) { red[wid][c] = s; red[wid][NF + c] = q; }
    }
    __syncthreads();
    if (threadIdx.x < 2 * NF) {
        float s = 0.f;
#pragma unroll
        for (int w = 0; w < 8; ++w) s += red[w][threadIdx.x];
        partial[(size_t)blockIdx.x * (2 * NF) + threadIdx.x] = s;
    }
}

// ---------------------------------------------------------------------------
// Kernel 6: BN finalize + normalize + transpose, fused.
// Each block redundantly reduces the NBLK partial records (deterministic,
// ~2.5K FMAs) into per-channel scale/shift in LDS, then normalizes its 256
// rows and writes the next-layer f16 B-matrix Ht[b][f][n].
// ---------------------------------------------------------------------------
__global__ void __launch_bounds__(256)
bn_normalize(const float* __restrict__ Hc, const float* __restrict__ partial,
             const float* __restrict__ g, const float* __restrict__ be,
             _Float16* __restrict__ Ht) {
    __shared__ float ab[2 * NF];   // [0..19]=scale, [20..39]=shift
    if (threadIdx.x < NF) {
        int c = threadIdx.x;
        float s = 0.f, q = 0.f;
        for (int i = 0; i < NBLK; ++i) {
            s += partial[(size_t)i * (2 * NF) + c];
            q += partial[(size_t)i * (2 * NF) + NF + c];
        }
        const float inv = 1.f / (float)ROWS;
        float mu  = s * inv;
        float var = q * inv - mu * mu;
        float rs  = rsqrtf(var + 1e-5f);
        float a   = g[c] * rs;
        ab[c]      = a;
        ab[NF + c] = be[c] - mu * a;
    }
    __syncthreads();

    int row = blockIdx.x * blockDim.x + threadIdx.x;   // b*512 + n
    int b = row >> 9;
    int n = row & 511;
    const float* hcr = Hc + (size_t)row * NF;
    _Float16* base = Ht + (size_t)b * HTF * NN + n;
#pragma unroll
    for (int c = 0; c < NF; ++c)
        base[(size_t)c * NN] = (_Float16)(hcr[c] * ab[c] + ab[NF + c]);
}

// ---------------------------------------------------------------------------
// Kernel 7: final readout: out[row] = dot(Hm[row, :120], fc_w) + fc_b
// ---------------------------------------------------------------------------
__global__ void __launch_bounds__(256)
final_fc(const float* __restrict__ Hm, const float* __restrict__ fcw,
         const float* __restrict__ fcb, float* __restrict__ out) {
    int row = blockIdx.x * blockDim.x + threadIdx.x;
    const float* hr = Hm + (size_t)row * (JJ * NF);
    float s = fcb[0];
    for (int k = 0; k < JJ * NF; ++k) s += hr[k] * fcw[k];
    out[row] = s;
}

// ---------------------------------------------------------------------------
extern "C" void kernel_launch(void* const* d_in, const int* in_sizes, int n_in,
                              void* d_out, int out_size, void* d_ws, size_t ws_size,
                              hipStream_t stream) {
    const float* W      = (const float*)d_in[0];
    const float* x      = (const float*)d_in[1];
    const float* fc1_w0 = (const float*)d_in[2];
    const float* fc1_b0 = (const float*)d_in[3];
    const float* fc2_w0 = (const float*)d_in[4];
    const float* fc2_b0 = (const float*)d_in[5];
    const float* gamma0 = (const float*)d_in[6];
    const float* beta0  = (const float*)d_in[7];
    const float* fc1_w  = (const float*)d_in[8];
    const float* fc1_b  = (const float*)d_in[9];
    const float* fc2_w  = (const float*)d_in[10];
    const float* fc2_b  = (const float*)d_in[11];
    const float* gamma  = (const float*)d_in[12];
    const float* beta   = (const float*)d_in[13];
    const float* fc_w   = (const float*)d_in[14];
    const float* fc_b   = (const float*)d_in[15];
    float* out = (float*)d_out;

    // ---- workspace carve-up (all 256B aligned) ----
    char* ws = (char*)d_ws;
    size_t off = 0;
    _Float16* Wh = (_Float16*)(ws + off); off += (size_t)BB * JJ * NN * NN * 2;  // 100.7 MB
    _Float16* Ht = (_Float16*)(ws + off); off += (size_t)BB * HTF * NN * 2;      // 1 MB
    float*    Hm = (float*)   (ws + off); off += (size_t)ROWS * JJ * NF * 4;     // 7.9 MB (also holds layer-0 Y)
    float*    Hc = (float*)   (ws + off); off += (size_t)ROWS * NF * 4;          // 1.3 MB
    float* partial = (float*) (ws + off); off += (size_t)NBLK * 2 * NF * 4;

    // 1) one-time (per call) HBM pass: f32 W -> f16 Wh, L2-resident thereafter
    repack_w<<<(BB * NN * NN) / 256, 256, 0, stream>>>(W, Wh);
    zero_pad<<<(BB * (HTF - NF) * NN) / 256, 256, 0, stream>>>(Ht);

    // 2) layer 0 (F=1)
    gmul0<<<(BB * JJ * NN) / 256, 256, 0, stream>>>(Wh, x, Hm);
    fc_bn<<<NBLK, 256, 0, stream>>>(Hm, JJ, fc1_w0, fc1_b0, fc2_w0, fc2_b0, Hc, partial);
    bn_normalize<<<NBLK, 256, 0, stream>>>(Hc, partial, gamma0, beta0, Ht);

    // 3) 20 middle layers: WMMA GEMM + FC + fused BN
    for (int L = 0; L < NL; ++L) {
        gemm_wmma<<<BB * JJ * (NN / 64), 128, 0, stream>>>(Wh, Ht, Hm);
        fc_bn<<<NBLK, 256, 0, stream>>>(Hm, JJ * NF,
                                        fc1_w + (size_t)L * 10 * JJ * NF,
                                        fc1_b + (size_t)L * 10,
                                        fc2_w + (size_t)L * 10 * JJ * NF,
                                        fc2_b + (size_t)L * 10,
                                        Hc, partial);
        bn_normalize<<<NBLK, 256, 0, stream>>>(Hc, partial,
                                               gamma + (size_t)L * NF,
                                               beta  + (size_t)L * NF, Ht);
    }

    // 4) final gmul + linear readout
    gemm_wmma<<<BB * JJ * (NN / 64), 128, 0, stream>>>(Wh, Ht, Hm);
    final_fc<<<ROWS / 256, 256, 0, stream>>>(Hm, fc_w, fc_b, out);
}
